// GAT_flatten_emp_50835232916298
// MI455X (gfx1250) — matrix operand order repaired and checked by reading, not compiled, not run on hardware
//
#include <hip/hip_runtime.h>

// ---------------------------------------------------------------------------
// 2-layer GAT for MI455X (gfx1250, wave32).
//  - Dense x@W GEMMs via v_wmma_f32_16x16x32_f16 (f16 in, f32 accum).
//    LDS-staged tiles, padded + transposed so every fragment load is a
//    bank-conflict-free ds_load_b128. 32x64 block tile, 2 WMMA accumulators
//    per wave sharing each B fragment (16 WMMAs per staged strip).
//  - Edge softmax via atomic-max / atomic-add passes (stats live in L2).
//  - Scatter aggregation: one wave per (edge, head); z[src] gathers are hot
//    in the 192MB L2, scatters ride global_atomic_add_f32.
// ---------------------------------------------------------------------------

typedef __attribute__((ext_vector_type(16))) _Float16 v16h;
typedef __attribute__((ext_vector_type(8)))  _Float16 v8h;
typedef __attribute__((ext_vector_type(8)))  float    v8f;

#define GK  256          // inner (K) dimension for both layer GEMMs
#define LDP (GK + 8)     // padded LDS row stride (528B: 16B-aligned, no bank conflicts)

// ---------------------------------------------------------------------------
// C[M x NC] = A[M x GK] * B[GK x NC], fp32 in/out, f16 WMMA compute.
// Block: 128 threads = 4 waves. Block tile: 32 rows x 64 cols.
// Each wave owns one 16-col strip and two 16-row tiles (shared B fragment).
// grid.x = ceil(M/32), grid.y = NC/64.
// ---------------------------------------------------------------------------
__global__ __launch_bounds__(128) void gat_gemm_wmma(
    const float* __restrict__ A, const float* __restrict__ B,
    float* __restrict__ C, int M, int NC) {
  __shared__ _Float16 sA[32 * LDP];        // 32 rows  x 256 K   (padded)
  __shared__ _Float16 sB[4][16 * LDP];     // 16 cols  x 256 K   (padded, col-major)

  const int tid  = threadIdx.x;
  const int wave = tid >> 5;
  const int lane = tid & 31;
  const int m0   = blockIdx.x * 32;
  const int n0   = blockIdx.y * 64 + wave * 16;

  // Stage A strip (32 x 256), fp32 -> f16, coalesced global reads.
  for (int i = tid; i < 32 * GK; i += 128) {
    const int r = i >> 8;              // GK == 256
    const int k = i & (GK - 1);
    const int gr = m0 + r;
    const float v = (gr < M) ? A[(size_t)gr * GK + k] : 0.0f;
    sA[r * LDP + k] = (_Float16)v;
  }
  // Stage this wave's B strip (256 x 16) transposed to column-major:
  // global reads coalesced over n, LDS writes land in distinct banks (pad).
  for (int i = lane; i < GK * 16; i += 32) {
    const int k = i >> 4;
    const int n = i & 15;
    sB[wave][n * LDP + k] = (_Float16)B[(size_t)k * NC + n0 + n];
  }
  __syncthreads();

  const int row = lane & 15;   // M row within tile (A) / N col (B,C)
  const int hi  = lane >> 4;   // half-wave selector per ISA layout tables

  v8f acc0 = {}, acc1 = {};
  union AV { v16h v; v8h half[2]; };

  for (int k0 = 0; k0 < GK; k0 += 32) {
    // A fragment (16x32 f16): lane m=row; halves hold K = hi*8 + 0..7 and
    // K = 16 + hi*8 + 0..7. Both 16B-aligned contiguous -> ds_load_b128 x2.
    const int ka = k0 + hi * 8;
    AV a0, a1, b;
    a0.half[0] = *(const v8h*)&sA[row * LDP + ka];
    a0.half[1] = *(const v8h*)&sA[row * LDP + ka + 16];
    a1.half[0] = *(const v8h*)&sA[(row + 16) * LDP + ka];
    a1.half[1] = *(const v8h*)&sA[(row + 16) * LDP + ka + 16];
    // B fragment (32x16 f16): lane n=row; halves hold K = hi*8 + 0..7 and
    // K = 16 + hi*8 + 0..7 in column-major sB -> ds_load_b128 x2.
    const int kb = k0 + hi * 8;
    b.half[0] = *(const v8h*)&sB[wave][row * LDP + kb];
    b.half[1] = *(const v8h*)&sB[wave][row * LDP + kb + 16];
    acc0 = __builtin_amdgcn_wmma_f32_16x16x32_f16(
        false, a0.v, false, b.v, (short)0, acc0, false, false);
    acc1 = __builtin_amdgcn_wmma_f32_16x16x32_f16(
        false, a1.v, false, b.v, (short)0, acc1, false, false);
  }

  // C/D layout: lanes 0-15 -> M = r, lanes 16-31 -> M = 8 + r; N = row.
  const int cn = n0 + row;
#pragma unroll
  for (int r = 0; r < 8; ++r) {
    const int cm0 = m0 + hi * 8 + r;
    if (cm0 < M) C[(size_t)cm0 * NC + cn] = acc0[r];
    const int cm1 = m0 + 16 + hi * 8 + r;
    if (cm1 < M) C[(size_t)cm1 * NC + cn] = acc1[r];
  }
}

// ---------------------------------------------------------------------------
// Per-(node, head) attention scores: el = <z, a_l>, er = <z, a_r>.
// One wave per (n, h); D-wide dot product with shuffle reduction.
// ---------------------------------------------------------------------------
__global__ void gat_scores(const float* __restrict__ z,
                           const float* __restrict__ al,
                           const float* __restrict__ ar,
                           float* __restrict__ el, float* __restrict__ er,
                           int Nn, int H, int D) {
  const int wid  = (blockIdx.x * blockDim.x + threadIdx.x) >> 5;
  const int lane = threadIdx.x & 31;
  if (wid >= Nn * H) return;
  const int n = wid / H;
  const int h = wid - n * H;
  const float* zr = z + (size_t)n * H * D + (size_t)h * D;
  float sl = 0.0f, sr = 0.0f;
  for (int d = lane; d < D; d += 32) {
    const float zv = zr[d];
    sl += zv * al[h * D + d];
    sr += zv * ar[h * D + d];
  }
#pragma unroll
  for (int off = 16; off >= 1; off >>= 1) {
    sl += __shfl_down(sl, off, 32);
    sr += __shfl_down(sr, off, 32);
  }
  if (lane == 0) { el[wid] = sl; er[wid] = sr; }
}

__global__ void gat_fill(float* __restrict__ p, float v, long long n) {
  const long long i = (long long)blockIdx.x * blockDim.x + threadIdx.x;
  if (i < n) p[i] = v;
}

// Order-independent float atomic max via sign-aware integer atomics.
__device__ inline void atomicMaxFloat(float* addr, float val) {
  if (val >= 0.0f) atomicMax((int*)addr, __float_as_int(val));
  else             atomicMin((unsigned int*)addr, __float_as_uint(val));
}

// Pass 1: raw edge score e = leaky_relu(el[src] + er[dst]); running max per dst.
__global__ void gat_edge_max(const float* __restrict__ el,
                             const float* __restrict__ er,
                             const int* __restrict__ src,
                             const int* __restrict__ dst,
                             float* __restrict__ ebuf, float* __restrict__ mbuf,
                             int Ee, int H) {
  const int idx = blockIdx.x * blockDim.x + threadIdx.x;
  if (idx >= Ee * H) return;
  const int e = idx / H;
  const int h = idx - e * H;
  float v = el[src[e] * H + h] + er[dst[e] * H + h];
  v = (v > 0.0f) ? v : 0.2f * v;   // leaky_relu, DGL default slope
  ebuf[idx] = v;
  atomicMaxFloat(&mbuf[dst[e] * H + h], v);
}

// Pass 2: ex = exp(e - max[dst]); running sum per dst.
__global__ void gat_edge_sum(float* __restrict__ ebuf,
                             const float* __restrict__ mbuf,
                             float* __restrict__ sbuf,
                             const int* __restrict__ dst, int Ee, int H) {
  const int idx = blockIdx.x * blockDim.x + threadIdx.x;
  if (idx >= Ee * H) return;
  const int e = idx / H;
  const int h = idx - e * H;
  const float ex = __expf(ebuf[idx] - mbuf[dst[e] * H + h]);
  ebuf[idx] = ex;
  atomicAdd(&sbuf[dst[e] * H + h], ex);
}

// Pass 3: agg[dst] += (ex / sum[dst]) * z[src].  One wave per (edge, head).
__global__ void gat_edge_aggregate(const float* __restrict__ z,
                                   const float* __restrict__ ebuf,
                                   const float* __restrict__ sbuf,
                                   const int* __restrict__ src,
                                   const int* __restrict__ dst,
                                   float* __restrict__ agg,
                                   int Ee, int H, int D) {
  const int wid  = (blockIdx.x * blockDim.x + threadIdx.x) >> 5;
  const int lane = threadIdx.x & 31;
  if (wid >= Ee * H) return;
  const int e = wid / H;
  const int h = wid - e * H;
  const int s = src[e], d = dst[e];
  const float alpha = ebuf[wid] / sbuf[d * H + h];
  const float* zr = z   + (size_t)s * H * D + (size_t)h * D;
  float*       ag = agg + (size_t)d * H * D + (size_t)h * D;
  for (int i = lane; i < D; i += 32)
    atomicAdd(&ag[i], alpha * zr[i]);
}

// Layer-1 epilogue: h = elu(agg + b), in place.
__global__ void gat_bias_elu(float* __restrict__ h, const float* __restrict__ b,
                             long long total, int F) {
  const long long i = (long long)blockIdx.x * blockDim.x + threadIdx.x;
  if (i >= total) return;
  const float v = h[i] + b[(int)(i % F)];
  h[i] = (v > 0.0f) ? v : (__expf(v) - 1.0f);
}

// Layer-2 epilogue: out = agg + b (H2 == 1, so head-mean is identity).
__global__ void gat_bias_add(float* __restrict__ o, const float* __restrict__ b,
                             long long total, int F) {
  const long long i = (long long)blockIdx.x * blockDim.x + threadIdx.x;
  if (i >= total) return;
  o[i] = o[i] + b[(int)(i % F)];
}

// ---------------------------------------------------------------------------
extern "C" void kernel_launch(void* const* d_in, const int* in_sizes, int n_in,
                              void* d_out, int out_size, void* d_ws, size_t ws_size,
                              hipStream_t stream) {
  const float* x   = (const float*)d_in[0];   // [N, 256]
  const float* W1  = (const float*)d_in[1];   // [256, 256]
  const float* al1 = (const float*)d_in[2];   // [4, 64]
  const float* ar1 = (const float*)d_in[3];   // [4, 64]
  const float* b1  = (const float*)d_in[4];   // [256]
  const float* W2  = (const float*)d_in[5];   // [256, 64]
  const float* al2 = (const float*)d_in[6];   // [1, 64]
  const float* ar2 = (const float*)d_in[7];   // [1, 64]
  const float* b2  = (const float*)d_in[8];   // [64]
  const int*   src = (const int*)d_in[9];     // [E]
  const int*   dst = (const int*)d_in[10];    // [E]
  float* out = (float*)d_out;                 // [N, 64]

  const int Nn = in_sizes[0] / 256;           // 50000
  const int Ee = in_sizes[9];                 // 800000
  const int H1 = 4, H2 = 1, D = 64;

  // Workspace layout (floats): z (reused z1/z2), agg/h1, el, er, max, sum, ebuf
  float* z    = (float*)d_ws;
  float* agg  = z    + (size_t)Nn * 256;
  float* el   = agg  + (size_t)Nn * 256;
  float* er   = el   + (size_t)Nn * H1;
  float* mbuf = er   + (size_t)Nn * H1;
  float* sbuf = mbuf + (size_t)Nn * H1;
  float* ebuf = sbuf + (size_t)Nn * H1;       // E * H1 floats
  (void)ws_size; (void)n_in; (void)out_size;

  const float NEG_INF = __builtin_huge_valf() * -1.0f;
  const int mb = (Nn + 31) / 32;

  // ---------------- Layer 1 ----------------
  gat_gemm_wmma<<<dim3(mb, 256 / 64), 128, 0, stream>>>(x, W1, z, Nn, 256);

  {
    const int waves = Nn * H1;
    gat_scores<<<(waves * 32 + 255) / 256, 256, 0, stream>>>(z, al1, ar1, el, er, Nn, H1, D);
  }
  gat_fill<<<((long long)Nn * H1 + 255) / 256, 256, 0, stream>>>(mbuf, NEG_INF, (long long)Nn * H1);
  gat_fill<<<((long long)Nn * H1 + 255) / 256, 256, 0, stream>>>(sbuf, 0.0f, (long long)Nn * H1);
  gat_fill<<<((long long)Nn * 256 + 255) / 256, 256, 0, stream>>>(agg, 0.0f, (long long)Nn * 256);

  {
    const int nt = Ee * H1;
    gat_edge_max<<<(nt + 255) / 256, 256, 0, stream>>>(el, er, src, dst, ebuf, mbuf, Ee, H1);
    gat_edge_sum<<<(nt + 255) / 256, 256, 0, stream>>>(ebuf, mbuf, sbuf, dst, Ee, H1);
    gat_edge_aggregate<<<((long long)nt * 32 + 255) / 256, 256, 0, stream>>>(
        z, ebuf, sbuf, src, dst, agg, Ee, H1, D);
  }
  gat_bias_elu<<<((long long)Nn * 256 + 255) / 256, 256, 0, stream>>>(agg, b1, (long long)Nn * 256, 256);

  // ---------------- Layer 2 ----------------
  gat_gemm_wmma<<<dim3(mb, 1), 128, 0, stream>>>(agg, W2, z, Nn, 64);

  {
    const int waves = Nn * H2;
    gat_scores<<<(waves * 32 + 255) / 256, 256, 0, stream>>>(z, al2, ar2, el, er, Nn, H2, D);
  }
  gat_fill<<<((long long)Nn * H2 + 255) / 256, 256, 0, stream>>>(mbuf, NEG_INF, (long long)Nn * H2);
  gat_fill<<<((long long)Nn * H2 + 255) / 256, 256, 0, stream>>>(sbuf, 0.0f, (long long)Nn * H2);
  gat_fill<<<((long long)Nn * 64 + 255) / 256, 256, 0, stream>>>(out, 0.0f, (long long)Nn * 64);

  {
    const int nt = Ee * H2;
    gat_edge_max<<<(nt + 255) / 256, 256, 0, stream>>>(el, er, src, dst, ebuf, mbuf, Ee, H2);
    gat_edge_sum<<<(nt + 255) / 256, 256, 0, stream>>>(ebuf, mbuf, sbuf, dst, Ee, H2);
    gat_edge_aggregate<<<((long long)nt * 32 + 255) / 256, 256, 0, stream>>>(
        z, ebuf, sbuf, src, dst, out, Ee, H2, D);
  }
  gat_bias_add<<<((long long)Nn * 64 + 255) / 256, 256, 0, stream>>>(out, b2, (long long)Nn * 64, 64);
}
